// GNN_6373731467381
// MI455X (gfx1250) — matrix operand order repaired and checked by reading, not compiled, not run on hardware
//
#include <hip/hip_runtime.h>
#include <hip/hip_bf16.h>

// ---------------------------------------------------------------------------
// Problem constants (from reference)
// ---------------------------------------------------------------------------
#define BSZ   2
#define NNOD  2048
#define ROWS  (BSZ * NNOD)      // 4096
#define HX    256
#define HL    64
#define HY    64
#define HE    256
#define HC    1344              // (3+1)*(256+64)+64
#define DIN   128               // BX*BXC
#define OUTE  2

// Input indices (jax tree_leaves: sorted dict keys, recursively).
#define I_E       0
#define I_X       1
#define I_LABEL   2
#define I_MASK    3
#define I_EMB     4
#define I_INX1_B  5
#define I_INX1_W  6
#define I_INX2_B  7
#define I_INX2_W  8
#define I_INY1_B  9
#define I_INY1_W  10
#define I_INY2_B  11
#define I_INY2_W  12
#define I_LAYER(l) (13 + 12*(l))   // +0 aggrX.b +1 aggrX.w +2 aggrl.b +3 aggrl.w
                                   // +4 lnX.b +5 lnX.g +6 lnl.b +7 lnl.g
                                   // +8 updX.b +9 updX.w +10 updl.b +11 updl.w
#define I_G1_B    49
#define I_G1_W    50
#define I_G2_B    51
#define I_G2_W    52
#define I_M1_B    53
#define I_M1_W    54
#define I_M2_B    55
#define I_M2_W    56
#define I_Y       57

typedef __attribute__((ext_vector_type(16))) __bf16 v16bf;
typedef __attribute__((ext_vector_type(2)))  __bf16 bf16x2;
typedef __attribute__((ext_vector_type(8)))  float  v8f;

// Packed fp32 -> bf16 conversion (lowers to v_cvt_pk_bf16_f32).
__device__ __forceinline__ unsigned int pk2bf(float lo, float hi) {
  bf16x2 p;
  p[0] = (__bf16)lo;
  p[1] = (__bf16)hi;
  return __builtin_bit_cast(unsigned int, p);
}

// LDS byte offset of a __shared__ pointer (generic -> AS3 -> int).
__device__ __forceinline__ unsigned int lds_off(const void* p) {
  return (unsigned int)(unsigned long long)
         (__attribute__((address_space(3))) const void*)p;
}

// ---------------------------------------------------------------------------
// B-operand prep: Bt[Npad][K] (bf16, zero-filled beyond N) from fp32 B[K][N].
// One thread per (n, k-pair).  Npad*K/2 threads, always a multiple of 256.
// ---------------------------------------------------------------------------
__global__ __launch_bounds__(256)
void k_prep_bt(const float* __restrict__ B, int ldb,
               unsigned int* __restrict__ Bt, int K, int N)
{
  int idx = blockIdx.x * blockDim.x + threadIdx.x;
  int kh = K >> 1;
  int kp = idx % kh;
  int n  = idx / kh;
  unsigned int v = 0;
  if (n < N) {
    const float* b0 = B + (size_t)(2 * kp) * ldb + n;
    v = pk2bf(b0[0], b0[ldb]);
  }
  Bt[(size_t)n * kh + kp] = v;
}

// ---------------------------------------------------------------------------
// GEMM: C[M,N] = act(A[M,K] @ B[K,N] + bias) * rowmask
//   B is pre-transposed bf16: Bt[Npad][K]; staged with
//   global_load_async_to_lds_b128 (ASYNCcnt).
//   ABF16=false : A fp32, converted via v_cvt_pk_bf16_f32 while staging
//   ABF16=true  : A bf16 in memory; also staged with async-to-LDS
// v_wmma_f32_16x16x32_bf16 inner product, fp32 accumulate.
// Block: 256 threads = 8 waves; tile 128x64; wave = 32x32
// (2 A-frags x 2 B-frags -> 4 WMMAs per K-step per wave).
// M multiple of 128, K multiple of 32; N bounds-guarded at stores only.
// ---------------------------------------------------------------------------
#define BM 128
#define BN 64
#define BK 32
#define LSTR 40   // LDS row stride in ushorts (32 + 8 pad, keeps 16B alignment)

template <bool ABF16>
__global__ __launch_bounds__(256)
void gemm_wmma(const void* __restrict__ Av, int lda,
               const unsigned short* __restrict__ Bt,   // [Npad][K] bf16
               const float* __restrict__ bias,
               const unsigned char* __restrict__ rowmask,
               float* __restrict__ C, int ldc,
               float* __restrict__ C2, int ldc2,
               int M, int N, int K, int relu)
{
  __shared__ alignas(16) unsigned short Asl[BM * LSTR];
  __shared__ alignas(16) unsigned short Bsl[BN * LSTR];

  const int tid  = threadIdx.x;
  const int lane = tid & 31;
  const int w    = tid >> 5;       // wave 0..7
  const int wm   = w >> 1;         // 0..3  (32-row group)
  const int wn   = w & 1;          // 0..1  (32-col group)
  const int half = lane >> 4;
  const int l15  = lane & 15;

  const int rowBase = blockIdx.y * BM;
  const int colBase = blockIdx.x * BN;

  // staging coordinates
  const int ar = tid >> 1;          // A: 0..127 rows, 16 elems/thread
  const int ac = (tid & 1) * 16;
  const int bn = tid >> 2;          // B: 0..63 rows (n), 8 bf16/thread
  const int bk = (tid & 3) * 8;

  const unsigned int dA = lds_off(&Asl[ar * LSTR + ac]);
  const unsigned int dB = lds_off(&Bsl[bn * LSTR + bk]);

  v8f acc00 = {}, acc01 = {}, acc10 = {}, acc11 = {};

  for (int k0 = 0; k0 < K; k0 += BK) {
    // ---- stage B: async DMA of pre-transposed bf16 tile --------------------
    {
      unsigned long long g = (unsigned long long)
          (Bt + (size_t)(colBase + bn) * K + (k0 + bk));
      asm volatile("global_load_async_to_lds_b128 %0, %1, off"
                   :: "v"(dB), "v"(g) : "memory");
    }

    // ---- stage A ----------------------------------------------------------
    if constexpr (ABF16) {
      const unsigned short* ab = (const unsigned short*)Av;
      unsigned long long g = (unsigned long long)
          (ab + (size_t)(rowBase + ar) * lda + (k0 + ac));
      asm volatile("global_load_async_to_lds_b128 %0, %1, off"
                   :: "v"(dA), "v"(g) : "memory");
      asm volatile("global_load_async_to_lds_b128 %0, %1, off offset:16"
                   :: "v"(dA), "v"(g) : "memory");
    } else {
      const float* ap = (const float*)Av + (size_t)(rowBase + ar) * lda + (k0 + ac);
      float4 a0 = ((const float4*)ap)[0];
      float4 a1 = ((const float4*)ap)[1];
      float4 a2 = ((const float4*)ap)[2];
      float4 a3 = ((const float4*)ap)[3];
      uint4 p0, p1;
      p0.x = pk2bf(a0.x, a0.y); p0.y = pk2bf(a0.z, a0.w);
      p0.z = pk2bf(a1.x, a1.y); p0.w = pk2bf(a1.z, a1.w);
      p1.x = pk2bf(a2.x, a2.y); p1.y = pk2bf(a2.z, a2.w);
      p1.z = pk2bf(a3.x, a3.y); p1.w = pk2bf(a3.z, a3.w);
      uint4* as = (uint4*)&Asl[ar * LSTR + ac];
      as[0] = p0; as[1] = p1;
      if (k0 + BK < K) __builtin_prefetch(ap + BK, 0, 0);
    }

    asm volatile("s_wait_asynccnt 0x0" ::: "memory");
    __syncthreads();

    // ---- fragments --------------------------------------------------------
    union Pack { uint4 u[2]; v16bf v; };
    const uint4* ar0 = (const uint4*)&Asl[(wm * 32 + l15) * LSTR];
    const uint4* ar1 = (const uint4*)&Asl[(wm * 32 + 16 + l15) * LSTR];
    Pack af0; af0.u[0] = ar0[half]; af0.u[1] = ar0[2 + half];
    Pack af1; af1.u[0] = ar1[half]; af1.u[1] = ar1[2 + half];
    const uint4* br0 = (const uint4*)&Bsl[(wn * 32 + l15) * LSTR];
    const uint4* br1 = (const uint4*)&Bsl[(wn * 32 + 16 + l15) * LSTR];
    Pack bf0; bf0.u[0] = br0[half]; bf0.u[1] = br0[2 + half];
    Pack bf1; bf1.u[0] = br1[half]; bf1.u[1] = br1[2 + half];

    acc00 = __builtin_amdgcn_wmma_f32_16x16x32_bf16(false, af0.v, false, bf0.v,
                                                    (short)0, acc00, false, false);
    acc01 = __builtin_amdgcn_wmma_f32_16x16x32_bf16(false, af0.v, false, bf1.v,
                                                    (short)0, acc01, false, false);
    acc10 = __builtin_amdgcn_wmma_f32_16x16x32_bf16(false, af1.v, false, bf0.v,
                                                    (short)0, acc10, false, false);
    acc11 = __builtin_amdgcn_wmma_f32_16x16x32_bf16(false, af1.v, false, bf1.v,
                                                    (short)0, acc11, false, false);
    __syncthreads();
  }

  // ---- epilogue: bias / relu / row-mask, dual-destination store -----------
  const int n0 = colBase + wn * 32 + l15;
  const int n1 = n0 + 16;
  const float bb0 = (bias && n0 < N) ? bias[n0] : 0.0f;
  const float bb1 = (bias && n1 < N) ? bias[n1] : 0.0f;
#pragma unroll
  for (int mi = 0; mi < 2; ++mi) {
    const v8f& a0 = mi ? acc10 : acc00;
    const v8f& a1 = mi ? acc11 : acc01;
#pragma unroll
    for (int r = 0; r < 8; ++r) {
      int m = rowBase + wm * 32 + mi * 16 + r + 8 * half;
      float mrow = rowmask ? (rowmask[m] ? 1.0f : 0.0f) : 1.0f;
      if (n0 < N) {
        float v = a0[r] + bb0;
        if (relu) v = fmaxf(v, 0.0f);
        v *= mrow;
        C[(size_t)m * ldc + n0] = v;
        if (C2) C2[(size_t)m * ldc2 + n0] = v;
      }
      if (n1 < N) {
        float v = a1[r] + bb1;
        if (relu) v = fmaxf(v, 0.0f);
        v *= mrow;
        C[(size_t)m * ldc + n1] = v;
        if (C2) C2[(size_t)m * ldc2 + n1] = v;
      }
    }
  }
}

// ---------------------------------------------------------------------------
// dinv[row] = rsqrt(max(1 + sum_j adj[row][j], 1));  one wave per row
// ---------------------------------------------------------------------------
__global__ __launch_bounds__(256)
void k_degree(const float* __restrict__ E, float* __restrict__ dinv)
{
  int wid  = (int)((blockIdx.x * blockDim.x + threadIdx.x) >> 5);
  int lane = threadIdx.x & 31;
  const float* row = E + (size_t)wid * NNOD * 2;
  float s = 0.0f;
  for (int j = lane; j < NNOD; j += 32) s += row[2 * j + 1];
#pragma unroll
  for (int o = 16; o > 0; o >>= 1) s += __shfl_xor(s, o, 32);
  if (lane == 0) dinv[wid] = rsqrtf(fmaxf(s + 1.0f, 1.0f));
}

// An_bf16[b][i][j] = bf16(dinv[b,i]*dinv[b,j]*(adj[b,i,j] + (i==j)))
__global__ __launch_bounds__(256)
void k_build_an_bf(const float* __restrict__ E, const float* __restrict__ dinv,
                   unsigned int* __restrict__ An2)
{
  size_t p = (size_t)blockIdx.x * blockDim.x + threadIdx.x;
  size_t idx = p * 2;
  int j = (int)(idx & (NNOD - 1));
  int i = (int)((idx >> 11) & (NNOD - 1));
  int b = (int)(idx >> 22);
  float di = dinv[b * NNOD + i];
  float v0 = di * dinv[b * NNOD + j]     * (E[idx * 2 + 1] + ((i == j)     ? 1.0f : 0.0f));
  float v1 = di * dinv[b * NNOD + j + 1] * (E[idx * 2 + 3] + ((i == j + 1) ? 1.0f : 0.0f));
  An2[p] = pk2bf(v0, v1);
}

// lb0 = emb[label+1] * mask  -> H slab (col 1024) and XL slab (col 256)
__global__ __launch_bounds__(256)
void k_embed(const int* __restrict__ label, const unsigned char* __restrict__ mask,
             const float* __restrict__ emb, float* __restrict__ H,
             float* __restrict__ XL)
{
  int idx = blockIdx.x * blockDim.x + threadIdx.x;  // ROWS*HL
  int h = idx & (HL - 1);
  int row = idx >> 6;
  float m = mask[row] ? 1.0f : 0.0f;
  float v = emb[(label[row] + 1) * HL + h] * m;
  H [(size_t)row * HC  + 1024 + h] = v;
  XL[(size_t)row * 320 + 256  + h] = v;
}

// tiny y MLP: (2,16)->64 relu ->64 relu ; one block of 128 threads
__global__ __launch_bounds__(128)
void k_y_mlp(const float* __restrict__ y,
             const float* __restrict__ w1, const float* __restrict__ b1,
             const float* __restrict__ w2, const float* __restrict__ b2,
             float* __restrict__ yvec)
{
  __shared__ float h1[BSZ][HY];
  int t = threadIdx.x, b = t >> 6, j = t & 63;
  float s = b1[j];
  for (int i = 0; i < 16; ++i) s += y[b * 16 + i] * w1[i * HY + j];
  h1[b][j] = fmaxf(s, 0.0f);
  __syncthreads();
  float s2 = b2[j];
  for (int i = 0; i < HY; ++i) s2 += h1[b][i] * w2[i * HY + j];
  yvec[b * HY + j] = fmaxf(s2, 0.0f);
}

// broadcast y to H (col 1280) and U (col 320) for every node row
__global__ __launch_bounds__(256)
void k_bcast_y(const float* __restrict__ yvec, float* __restrict__ H,
               float* __restrict__ U)
{
  int idx = blockIdx.x * blockDim.x + threadIdx.x;  // ROWS*HY
  int h = idx & (HY - 1);
  int row = idx >> 6;
  int b = row >> 11;
  float v = yvec[b * HY + h];
  H[(size_t)row * HC  + 1280 + h] = v;
  U[(size_t)row * 384 + 320  + h] = v;
}

// LayerNorm over last dim D (one wave per row); masked dual-destination store
template <int D>
__global__ __launch_bounds__(256)
void k_layernorm(const float* __restrict__ X,
                 const float* __restrict__ g, const float* __restrict__ be,
                 const unsigned char* __restrict__ mask,
                 float* __restrict__ out1, int ld1,
                 float* __restrict__ out2, int ld2)
{
  constexpr int CNT = D / 32;
  int wid  = (int)((blockIdx.x * blockDim.x + threadIdx.x) >> 5);
  int lane = threadIdx.x & 31;
  const float* x = X + (size_t)wid * D;
  float vals[CNT];
  float s = 0.0f;
#pragma unroll
  for (int c = 0; c < CNT; ++c) { vals[c] = x[lane + 32 * c]; s += vals[c]; }
#pragma unroll
  for (int o = 16; o > 0; o >>= 1) s += __shfl_xor(s, o, 32);
  float mean = s * (1.0f / D);
  float v = 0.0f;
#pragma unroll
  for (int c = 0; c < CNT; ++c) { float d = vals[c] - mean; v += d * d; }
#pragma unroll
  for (int o = 16; o > 0; o >>= 1) v += __shfl_xor(v, o, 32);
  float rstd = rsqrtf(v * (1.0f / D) + 1e-5f);
  float m = mask[wid] ? 1.0f : 0.0f;
#pragma unroll
  for (int c = 0; c < CNT; ++c) {
    int col = lane + 32 * c;
    float o_ = ((vals[c] - mean) * rstd * g[col] + be[col]) * m;
    out1[(size_t)wid * ld1 + col] = o_;
    out2[(size_t)wid * ld2 + col] = o_;
  }
}

// ---------------------------------------------------------------------------
// Host driver
// ---------------------------------------------------------------------------
static inline int npad64(int n) { return (n + 63) & ~63; }

static inline void prep_bt(hipStream_t s, const float* B, int ldb,
                           unsigned short* Bt, int K, int N)
{
  int total = npad64(N) * (K / 2);
  k_prep_bt<<<total / 256, 256, 0, s>>>(B, ldb, (unsigned int*)Bt, K, N);
}

static inline void gemm(hipStream_t s, const float* A, int lda,
                        const unsigned short* Bt, const float* bias,
                        const unsigned char* rmask,
                        float* C, int ldc, float* C2, int ldc2,
                        int M, int N, int K, int relu)
{
  dim3 g(npad64(N) / BN, M / BM);
  gemm_wmma<false><<<g, 256, 0, s>>>(A, lda, Bt, bias, rmask, C, ldc, C2, ldc2,
                                     M, N, K, relu);
}

static inline void gemm_abf(hipStream_t s, const unsigned short* A, int lda,
                            const unsigned short* Bt, const float* bias,
                            float* C, int ldc, int M, int N, int K)
{
  dim3 g(npad64(N) / BN, M / BM);
  gemm_wmma<true><<<g, 256, 0, s>>>(A, lda, Bt, bias, nullptr, C, ldc,
                                    nullptr, 0, M, N, K, 0);
}

extern "C" void kernel_launch(void* const* d_in, const int* in_sizes, int n_in,
                              void* d_out, int out_size, void* d_ws, size_t ws_size,
                              hipStream_t stream)
{
  (void)in_sizes; (void)n_in; (void)out_size; (void)ws_size;
  auto F  = [&](int i) { return (const float*)d_in[i]; };
  const float*         E     = F(I_E);
  const float*         Xin   = F(I_X);              // (4096,128) contiguous
  const int*           label = (const int*)d_in[I_LABEL];
  const unsigned char* mask  = (const unsigned char*)d_in[I_MASK];
  const float*         emb   = F(I_EMB);
  const float*         yin   = F(I_Y);

  // ---- workspace carve (256B aligned) ----
  char* base = (char*)d_ws;
  size_t off = 0;
  auto alloc = [&](size_t bytes) -> void* {
    void* p = base + off;
    off = (off + bytes + 255) & ~(size_t)255;
    return p;
  };
  unsigned short* An  = (unsigned short*)alloc((size_t)BSZ * NNOD * NNOD * 2); // bf16
  unsigned short* Btw = (unsigned short*)alloc((size_t)HC * HC * 2);   // weight Bt scratch
  unsigned short* BtP = (unsigned short*)alloc((size_t)BSZ * HX * NNOD * 2);
  unsigned short* BtPl= (unsigned short*)alloc((size_t)BSZ * HL * NNOD * 2);
  float* H    = (float*)alloc((size_t)ROWS * HC * 4);   // [X0..X3|l0..l3|y]
  float* G    = (float*)alloc((size_t)ROWS * HC * 4);
  float* XL   = (float*)alloc((size_t)ROWS * 320 * 4);  // [X|lb] working
  float* U    = (float*)alloc((size_t)ROWS * 384 * 4);  // [Xa|la|y]
  float* P    = (float*)alloc((size_t)ROWS * HX * 4);
  float* Pl   = (float*)alloc((size_t)ROWS * HL * 4);
  float* T    = (float*)alloc((size_t)ROWS * HX * 4);
  float* Tl   = (float*)alloc((size_t)ROWS * HL * 4);
  float* H2   = (float*)alloc((size_t)ROWS * HE * 4);
  float* dinv = (float*)alloc((size_t)ROWS * 4);
  float* yvec = (float*)alloc((size_t)BSZ * HY * 4);

  // ---- adjacency normalization (An built directly in bf16) ----
  k_degree<<<ROWS * 32 / 256, 256, 0, stream>>>(E, dinv);
  k_build_an_bf<<<(unsigned)((size_t)BSZ * NNOD * NNOD / 2 / 256), 256, 0, stream>>>(
      E, dinv, (unsigned int*)An);

  // ---- y MLP + input MLP + embedding ----
  k_y_mlp<<<1, 128, 0, stream>>>(yin, F(I_INY1_W), F(I_INY1_B),
                                 F(I_INY2_W), F(I_INY2_B), yvec);
  // T = relu(Xin @ W_inX1 + b)
  prep_bt(stream, F(I_INX1_W), HX, Btw, DIN, HX);
  gemm(stream, Xin, DIN, Btw, F(I_INX1_B), nullptr,
       T, HX, nullptr, 0, ROWS, HX, DIN, 1);
  // X0 = relu(T @ W_inX2 + b) * mask  -> XL[:,0:256] and H[:,0:256]
  prep_bt(stream, F(I_INX2_W), HX, Btw, HX, HX);
  gemm(stream, T, HX, Btw, F(I_INX2_B), mask,
       XL, 320, H, HC, ROWS, HX, HX, 1);
  k_embed  <<<ROWS * HL / 256, 256, 0, stream>>>(label, mask, emb, H, XL);
  k_bcast_y<<<ROWS * HY / 256, 256, 0, stream>>>(yvec, H, U);

  // ---- GCN layers ----
  for (int l = 0; l < 3; ++l) {
    const int LB = I_LAYER(l);
    const float* aggrX_b = F(LB + 0); const float* aggrX_w = F(LB + 1);
    const float* aggrl_b = F(LB + 2); const float* aggrl_w = F(LB + 3);
    const float* lnX_b   = F(LB + 4); const float* lnX_g   = F(LB + 5);
    const float* lnl_b   = F(LB + 6); const float* lnl_g   = F(LB + 7);
    const float* updX_b  = F(LB + 8); const float* updX_w  = F(LB + 9);
    const float* updl_b  = F(LB +10); const float* updl_w  = F(LB +11);

    // P  = [X|lb] @ W_aggrX   (no bias; GCN bias goes after An)
    prep_bt(stream, aggrX_w, HX, Btw, 320, HX);
    gemm(stream, XL, 320, Btw, nullptr, nullptr,
         P, HX, nullptr, 0, ROWS, HX, 320, 0);
    // Pl = lb @ W_aggrl
    prep_bt(stream, aggrl_w, HL, Btw, HL, HL);
    gemm(stream, XL + 256, 320, Btw, nullptr, nullptr,
         Pl, HL, nullptr, 0, ROWS, HL, HL, 0);
    // pre-transpose P/Pl per batch for the An GEMMs (both operands bf16+async)
    for (int b = 0; b < BSZ; ++b) {
      prep_bt(stream, P  + (size_t)b * NNOD * HX, HX,
              BtP  + (size_t)b * HX * NNOD, NNOD, HX);
      prep_bt(stream, Pl + (size_t)b * NNOD * HL, HL,
              BtPl + (size_t)b * HL * NNOD, NNOD, HL);
    }
    for (int b = 0; b < BSZ; ++b) {
      const unsigned short* Anb = An + (size_t)b * NNOD * NNOD;
      // Xa = An @ P + b_aggrX  -> U[:,0:256]
      gemm_abf(stream, Anb, NNOD, BtP + (size_t)b * HX * NNOD, aggrX_b,
               U + (size_t)b * NNOD * 384, 384, NNOD, HX, NNOD);
      // la = An @ Pl + b_aggrl -> U[:,256:320]
      gemm_abf(stream, Anb, NNOD, BtPl + (size_t)b * HL * NNOD, aggrl_b,
               U + (size_t)b * NNOD * 384 + 256, 384, NNOD, HL, NNOD);
    }
    // T = relu([Xa|la|y] @ W_updX + b)
    prep_bt(stream, updX_w, HX, Btw, 384, HX);
    gemm(stream, U, 384, Btw, updX_b, nullptr,
         T, HX, nullptr, 0, ROWS, HX, 384, 1);
    // X_{l+1} = LN(T)*mask -> H[:, 256*(l+1)] and XL[:,0:256]
    k_layernorm<HX><<<ROWS * 32 / 256, 256, 0, stream>>>(
        T, lnX_g, lnX_b, mask, H + 256 * (l + 1), HC, XL, 320);
    // Tl = relu(la @ W_updl + b)
    prep_bt(stream, updl_w, HL, Btw, HL, HL);
    gemm(stream, U + 256, 384, Btw, updl_b, nullptr,
         Tl, HL, nullptr, 0, ROWS, HL, HL, 1);
    // lb_{l+1} = LN(Tl)*mask -> H[:, 1024+64*(l+1)] and XL[:,256:320]
    k_layernorm<HL><<<ROWS * 32 / 256, 256, 0, stream>>>(
        Tl, lnl_g, lnl_b, mask, H + 1024 + 64 * (l + 1), HC, XL + 256, 320);
  }

  // ---- output head ----
  prep_bt(stream, F(I_G1_W), HC, Btw, HC, HC);
  gemm(stream, H, HC, Btw, F(I_G1_B), nullptr,
       G, HC, nullptr, 0, ROWS, HC, HC, 1);                 // relu
  prep_bt(stream, F(I_G2_W), HE, Btw, HC, HE);
  gemm(stream, G, HC, Btw, F(I_G2_B), nullptr,
       H2, HE, nullptr, 0, ROWS, HE, HC, 0);
  prep_bt(stream, F(I_M1_W), HE, Btw, HE, HE);
  gemm(stream, H2, HE, Btw, F(I_M1_B), nullptr,
       T, HE, nullptr, 0, ROWS, HE, HE, 1);                 // relu
  prep_bt(stream, F(I_M2_W), OUTE, Btw, HE, OUTE);
  gemm(stream, T, HE, Btw, F(I_M2_B), nullptr,
       (float*)d_out, OUTE, nullptr, 0, ROWS, OUTE, HE, 0);
}
